// CO2_27049704030994
// MI455X (gfx1250) — compile-verified, matrix operand order checked by reading, not run on hardware
//
#include <hip/hip_runtime.h>
#include <hip/hip_bf16.h>

typedef __attribute__((ext_vector_type(16))) __bf16 v16bf;
typedef __attribute__((ext_vector_type(8)))  __bf16 v8bf;
typedef __attribute__((ext_vector_type(8)))  float  v8f;

#define LEAKY_SLOPE 0.2f
#define CDNA5_ASYNC 1   // use global_load_async_to_lds_b128 (ASYNCcnt path)

static __device__ __forceinline__ float  bf2f(__bf16 x) { return (float)x; }
static __device__ __forceinline__ __bf16 f2bf(float x)  { return (__bf16)x; }

static __device__ __forceinline__ v8bf zero8() {
  v8bf z = {f2bf(0.f), f2bf(0.f), f2bf(0.f), f2bf(0.f),
            f2bf(0.f), f2bf(0.f), f2bf(0.f), f2bf(0.f)};
  return z;
}

// Convert 8 consecutive f32 (16B-aligned) to a v8bf.
static __device__ __forceinline__ v8bf cvt8(const float* __restrict__ src) {
  float4 a = *(const float4*)src;
  float4 b = *(const float4*)(src + 4);
  v8bf v = {f2bf(a.x), f2bf(a.y), f2bf(a.z), f2bf(a.w),
            f2bf(b.x), f2bf(b.y), f2bf(b.z), f2bf(b.w)};
  return v;
}

// Async DMA: copy 16B global -> LDS without touching VGPR data.
// LDS byte address = low 32 bits of the generic address (ISA aperture rule).
static __device__ __forceinline__ void copy16_g2l(const __bf16* __restrict__ g,
                                                  __bf16* l) {
#if CDNA5_ASYNC
  unsigned lds_addr = (unsigned)(uintptr_t)l;
  asm volatile("global_load_async_to_lds_b128 %0, %1, off"
               :: "v"(lds_addr), "v"(g)
               : "memory");
#else
  *(v8bf*)l = *(const v8bf*)g;
#endif
}

static __device__ __forceinline__ void async_wait() {
#if CDNA5_ASYNC
  asm volatile("s_wait_asynccnt 0" ::: "memory");
#endif
}

// Build one 16x32 bf16 WMMA operand fragment for this lane from an LDS tile row.
// Per ISA 7.12.2 (16-bit A/B layout): lane = h*16 + (m|n); elements e0..7 cover
// K = 8h..8h+7, e8..15 cover K = 16+8h..16+8h+7  -> two contiguous 16B chunks.
static __device__ __forceinline__ v16bf frag16(const __bf16* p, int h) {
  v8bf lo = *(const v8bf*)(p + h * 8);
  v8bf hi = *(const v8bf*)(p + 16 + h * 8);
  return __builtin_shufflevector(lo, hi, 0, 1, 2, 3, 4, 5, 6, 7,
                                 8, 9, 10, 11, 12, 13, 14, 15);
}

// Straight f32 -> bf16 conversion, 8 elements per thread (vectorized).
__global__ __launch_bounds__(256) void cvt_k(const float* __restrict__ src,
                                             __bf16* __restrict__ dst, int n8) {
  const int i = blockIdx.x * blockDim.x + threadIdx.x;
  if (i < n8) *(v8bf*)(dst + (size_t)i * 8) = cvt8(src + (size_t)i * 8);
}

// cls1_w [N, K, 3] f32  ->  tap-major [3, N, K] bf16 (makes conv1's B-side
// K-contiguous so it can use the async DMA path).
__global__ __launch_bounds__(256) void cvt_w1_k(const float* __restrict__ w1,
                                                __bf16* __restrict__ w1t,
                                                int N, int K) {
  const int tap = blockIdx.y;
  const int r   = blockIdx.x * blockDim.x + threadIdx.x;  // unit of 8 along d
  const int k8  = K >> 3;
  const int n   = r / k8;
  const int d   = (r - n * k8) * 8;
  const float* s = w1 + ((size_t)n * K + d) * 3 + tap;
  v8bf v = {f2bf(s[0]),  f2bf(s[3]),  f2bf(s[6]),  f2bf(s[9]),
            f2bf(s[12]), f2bf(s[15]), f2bf(s[18]), f2bf(s[21])};
  *(v8bf*)(w1t + ((size_t)tap * N + n) * K + d) = v;
}

// Generic tiled WMMA GEMM:  C = act( A * B^T * alpha + bias )
//   A: [M,K] bf16 row-major (optional row-gather via rowIdx; optional conv3
//      mode: virtual K=3*Kd with row shift tap-1)
//   B: bf16; normal: [N,K] row-major; conv3: tap-major [3,N,K]
//   C: [M,N] row-major (f32 or bf16), batched via blockIdx.z strides.
// 128x128 block tile, 8 wave32 waves (each 64x32 = 4x2 WMMA accumulators),
// K stepped by 32 through double-buffered LDS. All staging is async DMA
// (global_load_async_to_lds_b128): tile s+1 is issued right after the single
// per-step barrier and completes during tile-s WMMAs. Requires K % 32 == 0.
template <bool C3, bool IDX, bool BIAS, bool ACT, bool OBF>
__global__ __launch_bounds__(256) void gemm_k(
    const __bf16* __restrict__ A, const __bf16* __restrict__ Bm,
    const float* __restrict__ bias, void* __restrict__ Cp,
    int M, int N, int K, int Tconv, const int* __restrict__ rowIdx,
    long sAb, long sBb, long sCb, float alpha) {
  __shared__ __align__(16) __bf16 sA[2][128 * 40];
  __shared__ __align__(16) __bf16 sB[2][128 * 40];

  const int z = blockIdx.z;
  A  += (size_t)z * sAb;
  Bm += (size_t)z * sBb;
  float*  Cf = (float*)Cp;
  __bf16* Cb = (__bf16*)Cp;
  if (OBF) Cb += (size_t)z * sCb; else Cf += (size_t)z * sCb;

  const int tid  = threadIdx.x;
  const int lane = tid & 31;
  const int wave = tid >> 5;
  const int wm   = (wave >> 2) * 64;   // wave M offset within block tile
  const int wn   = (wave & 3) * 32;    // wave N offset within block tile
  const int lm   = lane & 15;
  const int h    = lane >> 4;
  const int bM   = blockIdx.y * 128;
  const int bN   = blockIdx.x * 128;
  const int Ktot   = C3 ? 3 * K : K;
  const int nsteps = Ktot >> 5;   // K-steps of 32

  int tmod0 = 0;
  if (C3) tmod0 = bM % Tconv;     // one scalar mod per block

  // Issue the async DMAs for one 128x32 A-tile + B-tile into buffer pb.
  // 512 chunks of 8 contiguous k-elements; 2 chunks per thread; all guards
  // are uniform per chunk, so each valid chunk is one 16B async transfer.
  auto issue = [&](int k0, int pb) {
#pragma unroll
    for (int it = 0; it < 2; ++it) {
      const int chunk = tid + it * 256;
      const int row   = chunk >> 2;        // 0..127
      const int c0    = (chunk & 3) * 8;   // 0,8,16,24
      const int kkb   = k0 + c0;
      const int gr    = bM + row;
      const int gc    = bN + row;
      __bf16* dA = &sA[pb][row * 40 + c0];
      __bf16* dB = &sB[pb][row * 40 + c0];

      int tap = 0, d0 = kkb;  // constant within a chunk (K multiple of 8)
      if (C3) {
        tap = (kkb >= 2 * K) ? 2 : ((kkb >= K) ? 1 : 0);
        d0  = kkb - tap * K;
      }

      // ---- A side ----
      bool aok  = (gr < M);
      int  arow = gr;
      if (C3) {
        if (aok) {
          int t = tmod0 + row;
          if (t >= Tconv) t -= Tconv;
          const int tt = t + tap - 1;
          aok  = (tt >= 0) && (tt < Tconv);
          arow = gr + tap - 1;
        }
      } else if (IDX) {
        if (aok) arow = rowIdx[gr];
      }
      if (aok) copy16_g2l(A + (size_t)arow * K + (C3 ? d0 : kkb), dA);
      else     *(v8bf*)dA = zero8();

      // ---- B side ----
      if (gc < N) {
        const __bf16* src = C3 ? (Bm + ((size_t)tap * N + gc) * K + d0)
                               : (Bm + (size_t)gc * K + kkb);
        copy16_g2l(src, dB);
      } else {
        *(v8bf*)dB = zero8();
      }
    }
  };

  v8f acc[4][2] = {};

  issue(0, 0);
  for (int s = 0; s < nsteps; ++s) {
    async_wait();        // my tile-s DMAs done
    __syncthreads();     // everyone's tile-s in LDS; prev compute reads done
    if (s + 1 < nsteps) issue((s + 1) * 32, (s + 1) & 1);  // overlap with WMMA

    const int pb = s & 1;
    v16bf aF[4], bF[2];
#pragma unroll
    for (int i = 0; i < 4; ++i)
      aF[i] = frag16(&sA[pb][(wm + i * 16 + lm) * 40], h);
#pragma unroll
    for (int j = 0; j < 2; ++j)
      bF[j] = frag16(&sB[pb][(wn + j * 16 + lm) * 40], h);
#pragma unroll
    for (int i = 0; i < 4; ++i)
#pragma unroll
      for (int j = 0; j < 2; ++j)
        acc[i][j] = __builtin_amdgcn_wmma_f32_16x16x32_bf16(
            false, aF[i], false, bF[j], (short)0, acc[i][j], false, false);
  }

  // ---- epilogue: scale, bias, LeakyReLU, store ----
  // C layout: lane l, vgpr r -> M_local = r + 8*(l>>4), N_local = l&15.
#pragma unroll
  for (int i = 0; i < 4; ++i) {
#pragma unroll
    for (int j = 0; j < 2; ++j) {
#pragma unroll
      for (int r = 0; r < 8; ++r) {
        const int grow = bM + wm + i * 16 + r + 8 * h;
        const int gcol = bN + wn + j * 16 + lm;
        if (grow < M && gcol < N) {
          float v = acc[i][j][r] * alpha;
          if (BIAS) v += bias[gcol];
          if (ACT)  v = (v > 0.f) ? v : LEAKY_SLOPE * v;
          if (OBF) Cb[(size_t)grow * N + gcol] = f2bf(v);
          else     Cf[(size_t)grow * N + gcol] = v;
        }
      }
    }
  }
}

// conv2 (1x1, N=21): one block per time-row; h row staged in LDS.
__global__ __launch_bounds__(128) void conv2_k(
    const __bf16* __restrict__ hbuf, const float* __restrict__ w2,
    const float* __restrict__ b2, float* __restrict__ out, int D, int NC1) {
  const int r = blockIdx.x;
  __shared__ float sh[2048];
  for (int d = threadIdx.x; d < D; d += blockDim.x)
    sh[d] = bf2f(hbuf[(size_t)r * D + d]);
  __syncthreads();
  const int c = threadIdx.x;
  if (c < NC1) {
    float s = b2[c];
    for (int d = 0; d < D; ++d) s += w2[(size_t)c * D + d] * sh[d];
    out[(size_t)r * NC1 + c] = s;
  }
}

// Exact top-k selection via rank counting (matches jax.lax.top_k's tie-break:
// earlier index wins on ties). Selection order is irrelevant downstream
// (attention + mean is permutation invariant). Emits global nfeat row ids.
__global__ __launch_bounds__(256) void topk_k(
    const float* __restrict__ x_atn, int* __restrict__ idxOut, int T, int KS) {
  const int b = blockIdx.x;
  __shared__ float sv[1536];
  for (int t = threadIdx.x; t < T; t += blockDim.x) sv[t] = x_atn[(size_t)b * T + t];
  __syncthreads();
  for (int t = threadIdx.x; t < T; t += blockDim.x) {
    const float val = sv[t];
    int rank = 0;
    for (int j = 0; j < T; ++j) {
      const float o = sv[j];
      rank += (o > val) || (o == val && j < t);
    }
    if (rank < KS) idxOut[b * KS + rank] = b * T + t;
  }
}

// Row softmax of scores then column mean: w[j] = mean_m softmax(S)[m][j].
__global__ __launch_bounds__(192) void softmax_colmean_k(
    const float* __restrict__ S, float* __restrict__ w, int Nn) {
  const int b = blockIdx.x;
  const float* Sb = S + (size_t)b * Nn * Nn;
  __shared__ float mx[192];
  __shared__ float inv[192];
  const int t = threadIdx.x;
  if (t < Nn) {
    float m = -3.4e38f;
    for (int j = 0; j < Nn; ++j) m = fmaxf(m, Sb[t * Nn + j]);
    float s = 0.f;
    for (int j = 0; j < Nn; ++j) s += __expf(Sb[t * Nn + j] - m);
    mx[t]  = m;
    inv[t] = 1.f / s;
  }
  __syncthreads();
  if (t < Nn) {
    float acc = 0.f;
    for (int m = 0; m < Nn; ++m) acc += __expf(Sb[m * Nn + t] - mx[m]) * inv[m];
    w[b * Nn + t] = acc / (float)Nn;
  }
}

// pooled[b,n] = 0.5 * ( sum_j w[b,j] * v[b,j,n]  +  mean_i nfeat[idx[b,i], n] )
__global__ __launch_bounds__(256) void pooled_k(
    const __bf16* __restrict__ vmat, const __bf16* __restrict__ nfeat,
    const int* __restrict__ idx, const float* __restrict__ w,
    float* __restrict__ pooled, int KS, int D) {
  const int b = blockIdx.y;
  const int n = blockIdx.x * blockDim.x + threadIdx.x;
  float s1 = 0.f, s2 = 0.f;
  for (int j = 0; j < KS; ++j)
    s1 += w[b * KS + j] * bf2f(vmat[(size_t)(b * KS + j) * D + n]);
  for (int i = 0; i < KS; ++i)
    s2 += bf2f(nfeat[(size_t)idx[b * KS + i] * D + n]);
  pooled[(size_t)b * D + n] = 0.5f * (s1 + s2 / (float)KS);
}

// Final classifier stage 1 on pooled: only the CENTER tap of cls1_w matters
// (k=3, pad=1 on a length-1 sequence). One block per output feature.
__global__ __launch_bounds__(256) void h2_k(
    const float* __restrict__ pooled, const float* __restrict__ w1,
    const float* __restrict__ b1, float* __restrict__ h2, int D, int B) {
  const int dout = blockIdx.x;
  __shared__ float sw[2048];
  for (int d = threadIdx.x; d < D; d += blockDim.x)
    sw[d] = w1[((size_t)dout * D + d) * 3 + 1];
  __syncthreads();
  const int b = threadIdx.x;
  if (b < B) {
    float s = b1[dout];
    for (int d = 0; d < D; ++d) s += sw[d] * pooled[(size_t)b * D + d];
    h2[(size_t)b * D + dout] = (s > 0.f) ? s : LEAKY_SLOPE * s;
  }
}

// logits = cls2(h2); drop background; EDL loss + uncertainty.
__global__ __launch_bounds__(512) void edl_k(
    const float* __restrict__ h2, const float* __restrict__ w2,
    const float* __restrict__ b2, const int* __restrict__ labels,
    float* __restrict__ out, int D, int NC1, int B, int outc) {
  __shared__ float lg[336];   // 16 * 21
  __shared__ float lred[16];
  const int t = threadIdx.x;
  if (t < B * NC1) {
    const int b = t / NC1, c = t - b * NC1;
    float s = b2[c];
    for (int d = 0; d < D; ++d) s += w2[(size_t)c * D + d] * h2[(size_t)b * D + d];
    lg[t] = s;
  }
  __syncthreads();
  if (t < B) {
    const int NC = NC1 - 1;
    float S = 0.f;
    for (int c = 0; c < NC; ++c) {
      float l = fminf(fmaxf(lg[t * NC1 + c], -10.f), 10.f);
      S += __expf(l) + 1.f;
    }
    const int lab = labels[t];
    float ll    = fminf(fmaxf(lg[t * NC1 + lab], -10.f), 10.f);
    float alpha = __expf(ll) + 1.f;
    lred[t] = __logf(S) - __logf(alpha);
    out[outc + 1 + t] = (float)NC / S;      // uncertainty
  }
  __syncthreads();
  if (t == 0) {
    float s = 0.f;
    for (int b = 0; b < B; ++b) s += lred[b];
    out[outc] = s / (float)B;               // mean loss
  }
}

extern "C" void kernel_launch(void* const* d_in, const int* in_sizes, int n_in,
                              void* d_out, int out_size, void* d_ws, size_t ws_size,
                              hipStream_t stream) {
  (void)in_sizes; (void)n_in; (void)out_size; (void)ws_size;
  const float* inputs  = (const float*)d_in[0];   // [16,1536,2048]
  const float* x_atn   = (const float*)d_in[1];   // [16,1536,1]
  const int*   labels  = (const int*)d_in[2];     // [16]
  const float* fusionW = (const float*)d_in[3];   // [2048,2048,1]
  const float* fusionB = (const float*)d_in[4];   // [2048]
  const float* cls1W   = (const float*)d_in[5];   // [2048,2048,3]
  const float* cls1B   = (const float*)d_in[6];   // [2048]
  const float* cls2W   = (const float*)d_in[7];   // [21,2048,1]
  const float* cls2B   = (const float*)d_in[8];   // [21]
  const float* Wq      = (const float*)d_in[9];   // [2048,2048]
  const float* Wk      = (const float*)d_in[10];
  const float* Wv      = (const float*)d_in[11];

  const int B = 16, T = 1536, D = 2048, NC1 = 21, KS = T / 8; // 192
  const int M  = B * T;    // 24576
  const int MS = B * KS;   // 3072

  char* ws = (char*)d_ws;
  size_t off = 0;
  auto wsalloc = [&](size_t bytes) -> void* {
    void* p = ws + off;
    off += (bytes + 255) & ~(size_t)255;
    return p;
  };
  __bf16* xbf    = (__bf16*)wsalloc((size_t)M * D * 2);     // inputs bf16 (dead after fusion)
  __bf16* nfeat  = (__bf16*)wsalloc((size_t)M * D * 2);     // fused features (bf16)
  __bf16* wfb    = (__bf16*)wsalloc((size_t)D * D * 2);     // fusion_w bf16
  __bf16* w1t    = (__bf16*)wsalloc((size_t)3 * D * D * 2); // cls1_w tap-major bf16
  __bf16* wqb    = (__bf16*)wsalloc((size_t)D * D * 2);
  __bf16* wkb    = (__bf16*)wsalloc((size_t)D * D * 2);
  __bf16* wvb    = (__bf16*)wsalloc((size_t)D * D * 2);
  __bf16* qb     = (__bf16*)wsalloc((size_t)MS * D * 2);
  __bf16* kb     = (__bf16*)wsalloc((size_t)MS * D * 2);
  __bf16* vb     = (__bf16*)wsalloc((size_t)MS * D * 2);
  float*  sc     = (float*)wsalloc((size_t)B * KS * KS * 4);
  float*  wcol   = (float*)wsalloc((size_t)B * KS * 4);
  int*    idx    = (int*)wsalloc((size_t)MS * 4);
  float*  pooled = (float*)wsalloc((size_t)B * D * 4);
  float*  h2     = (float*)wsalloc((size_t)B * D * 4);
  __bf16* hbuf   = xbf;   // reuse: conv1 output overwrites dead inputs-bf16

  float* out = (float*)d_out;
  const dim3 blk(256);

  // 0) one-time f32 -> bf16 weight/activation conversions
  cvt_k<<<dim3(M * D / 8 / 256), blk, 0, stream>>>(inputs, xbf, M * D / 8);
  cvt_k<<<dim3(D * D / 8 / 256), blk, 0, stream>>>(fusionW, wfb, D * D / 8);
  cvt_k<<<dim3(D * D / 8 / 256), blk, 0, stream>>>(Wq, wqb, D * D / 8);
  cvt_k<<<dim3(D * D / 8 / 256), blk, 0, stream>>>(Wk, wkb, D * D / 8);
  cvt_k<<<dim3(D * D / 8 / 256), blk, 0, stream>>>(Wv, wvb, D * D / 8);
  cvt_w1_k<<<dim3(D * D / 8 / 256, 3), blk, 0, stream>>>(cls1W, w1t, D, D);

  // 1) fusion 1x1 conv: nfeat = leaky(xbf @ wfb^T + b)   [24576x2048x2048]
  gemm_k<false, false, true, true, true>
      <<<dim3(D / 128, M / 128, 1), blk, 0, stream>>>(
          xbf, wfb, fusionB, nfeat, M, D, D, 0, nullptr, 0, 0, 0, 1.0f);

  // 2) conv1 (k=3, pad=1) as virtual-K GEMM: h = leaky(conv(nfeat))  [K=6144]
  gemm_k<true, false, true, true, true>
      <<<dim3(D / 128, M / 128, 1), blk, 0, stream>>>(
          nfeat, w1t, cls1B, hbuf, M, D, D, T, nullptr, 0, 0, 0, 1.0f);

  // 3) conv2 (1x1, 21 outputs) -> x_cls_t flat in d_out[0 .. 516095]
  conv2_k<<<dim3(M), dim3(128), 0, stream>>>(hbuf, cls2W, cls2B, out, D, NC1);

  // 4) exact top-k (k=192) on attention weights
  topk_k<<<dim3(B), dim3(256), 0, stream>>>(x_atn, idx, T, KS);

  // 5) q/k/v projections on gathered rows (row-indirect A)  [3072x2048x2048]
  gemm_k<false, true, false, false, true>
      <<<dim3(D / 128, MS / 128, 1), blk, 0, stream>>>(
          nfeat, wqb, nullptr, qb, MS, D, D, 0, idx, 0, 0, 0, 1.0f);
  gemm_k<false, true, false, false, true>
      <<<dim3(D / 128, MS / 128, 1), blk, 0, stream>>>(
          nfeat, wkb, nullptr, kb, MS, D, D, 0, idx, 0, 0, 0, 1.0f);
  gemm_k<false, true, false, false, true>
      <<<dim3(D / 128, MS / 128, 1), blk, 0, stream>>>(
          nfeat, wvb, nullptr, vb, MS, D, D, 0, idx, 0, 0, 0, 1.0f);

  // 6) attention scores, batched over B: S = (q @ k^T) / sqrt(D)
  const float scale = 0.022097086912079608f;  // 1/sqrt(2048)
  gemm_k<false, false, false, false, false>
      <<<dim3(2, 2, B), blk, 0, stream>>>(
          qb, kb, nullptr, sc, KS, KS, D, 0, nullptr,
          (long)KS * D, (long)KS * D, (long)KS * KS, scale);

  // 7) row softmax + column mean (mean over positions commutes with dist@v)
  softmax_colmean_k<<<dim3(B), dim3(192), 0, stream>>>(sc, wcol, KS);

  // 8) pooled = 0.5*(w^T v + mean(x_topk))
  pooled_k<<<dim3(D / 256, B), dim3(256), 0, stream>>>(vb, nfeat, idx, wcol,
                                                       pooled, KS, D);

  // 9) final classifier stage 1 (center tap only) on pooled
  h2_k<<<dim3(D), dim3(256), 0, stream>>>(pooled, cls1W, cls1B, h2, D, B);

  // 10) logits + EDL loss/uncertainty -> d_out[516096 ..]
  edl_k<<<dim3(1), dim3(512), 0, stream>>>(h2, cls2W, cls2B, labels, out, D,
                                           NC1, B, M * NC1);
}
